// FNetBlock_22058952032651
// MI455X (gfx1250) — compile-verified, matrix operand order checked by reading, not run on hardware
//
#include <hip/hip_runtime.h>
#include <hip/hip_bf16.h>

#define CH      128
#define GROUPS  8
#define CPG     16          // channels per group
#define HW      65536       // 256*256
#define EPS     1e-5f
#define TILE_PIX 128        // pixels per block (M)
#define LDA     136         // padded row pitch in halves (128 + 8 -> conflict-free frags)
#define LDO     132         // padded row pitch in floats for the output tile

typedef __attribute__((ext_vector_type(16))) _Float16 v16h;
typedef __attribute__((ext_vector_type(8)))  _Float16 v8h;
typedef __attribute__((ext_vector_type(8)))  float    v8f;

// ---------------------------------------------------------------- zero ws
__global__ void ws_zero_kernel(float* ws) {
    ws[threadIdx.x] = 0.0f;   // 128 floats: 64 x (sum, sumsq)
}

// ------------------------------------------------- pass 1: groupnorm stats
// grid = (64 bg-pairs, 64 chunks), block = 256. Each (b,g) covers a
// contiguous 16*65536 = 1M-float plane region.
__global__ __launch_bounds__(256)
void gn_stats_kernel(const float* __restrict__ x, float* __restrict__ ws) {
    const int bg    = blockIdx.x;   // b*8+g
    const int chunk = blockIdx.y;   // 0..63
    const float4* base =
        (const float4*)(x + (size_t)bg * (CPG * (size_t)HW) + (size_t)chunk * 16384);

    float s = 0.0f, s2 = 0.0f;
    #pragma unroll 4
    for (int i = threadIdx.x; i < 4096; i += 256) {   // 4096 float4 per block
        float4 v = base[i];
        s  += v.x + v.y + v.z + v.w;
        s2 += v.x * v.x + v.y * v.y + v.z * v.z + v.w * v.w;
    }
    __shared__ float rs[256], rq[256];
    rs[threadIdx.x] = s; rq[threadIdx.x] = s2;
    __syncthreads();
    for (int off = 128; off > 0; off >>= 1) {
        if (threadIdx.x < off) {
            rs[threadIdx.x] += rs[threadIdx.x + off];
            rq[threadIdx.x] += rq[threadIdx.x + off];
        }
        __syncthreads();
    }
    if (threadIdx.x == 0) {
        atomicAdd(&ws[2 * bg + 0], rs[0]);
        atomicAdd(&ws[2 * bg + 1], rq[0]);
    }
}

// ---------------------------------------------------------- fragment load
// A-frag (16x32 f16, MxK) for one k-chunk, per ISA layout:
//   lanes 0-15 : row M=lane,    K = {kc+0..7, kc+16..23}
//   lanes16-31 : row M=lane-16, K = {kc+8..15, kc+24..31}
// B-frag is the symmetric layout with N=lane and rows stored [n][k].
// p0 already points at (row, kc + hi*8); rows are 272B so 16B aligned.
__device__ __forceinline__ v16h load_frag(const _Float16* p0) {
    v8h lo = *(const v8h*)(p0);
    v8h hv = *(const v8h*)(p0 + 16);
    v16h r;
    #pragma unroll
    for (int i = 0; i < 8; ++i) { r[i] = lo[i]; r[8 + i] = hv[i]; }
    return r;
}

// ------------------------------------- pass 2: fused GN + GEMM+ReLU+GEMM
// grid = (512 pixel tiles, 8 batches), block = 256 (8 wave32).
__global__ __launch_bounds__(256)
void fused_gn_ffn_kernel(const float* __restrict__ x,
                         const float* __restrict__ gn_w,
                         const float* __restrict__ gn_b,
                         const float* __restrict__ w1,
                         const float* __restrict__ b1,
                         const float* __restrict__ w2,
                         const float* __restrict__ b2,
                         const float* __restrict__ ws,
                         float* __restrict__ out) {
    extern __shared__ __align__(16) char smem[];
    float*    scaleS = (float*)smem;                         //  512 B
    float*    shiftS = (float*)(smem + 512);                 //  512 B
    _Float16* ldsA   = (_Float16*)(smem + 1024);             // 128*136*2 = 34816 B (A, later H)
    _Float16* ldsW1  = ldsA  + TILE_PIX * LDA;               // 34816 B
    _Float16* ldsW2  = ldsW1 + CH * LDA;                     // 34816 B
    // After GEMM2 the A/W1 regions are dead; reuse them (69632 B) as an fp32
    // output tile [channel][pixel] (128*132*4 = 67584 B) for coalesced stores.
    float*    ldsO   = (float*)(smem + 1024);

    const int tid  = threadIdx.x;
    const int b    = blockIdx.y;
    const int pix0 = blockIdx.x * TILE_PIX;

    // per-channel scale/shift from group stats
    if (tid < CH) {
        const int   g   = tid >> 4;
        const float cnt = (float)(CPG * (size_t)HW);
        const float m   = ws[2 * (b * GROUPS + g) + 0] / cnt;
        const float var = ws[2 * (b * GROUPS + g) + 1] / cnt - m * m;
        const float inv = rsqrtf(var + EPS);
        const float sc  = gn_w[tid] * inv;
        scaleS[tid] = sc;
        shiftS[tid] = gn_b[tid] - m * sc;
    }
    __syncthreads();

    // ---- stage normalized x as f16 into ldsA[pixel][channel] (transposed scatter)
    const float* xb = x + (size_t)b * CH * (size_t)HW + pix0;
    for (int i = tid; i < CH * (TILE_PIX / 4); i += 256) {   // 4096 float4
        const int c  = i >> 5;
        const int p  = (i & 31) << 2;
        const float* src = xb + (size_t)c * HW + p;
        if ((i & 31) == 0) __builtin_prefetch(src, 0, 3);    // keep row near for epilogue re-read
        float4 v = *(const float4*)src;
        const float sc = scaleS[c], sh = shiftS[c];
        ldsA[(p + 0) * LDA + c] = (_Float16)(v.x * sc + sh);
        ldsA[(p + 1) * LDA + c] = (_Float16)(v.y * sc + sh);
        ldsA[(p + 2) * LDA + c] = (_Float16)(v.z * sc + sh);
        ldsA[(p + 3) * LDA + c] = (_Float16)(v.w * sc + sh);
    }
    // ---- stage weights as f16 [out][in] (contiguous rows)
    for (int i = tid; i < CH * (CH / 4); i += 256) {         // 4096 float4
        const int o  = i >> 5;
        const int c0 = (i & 31) << 2;
        float4 a = *(const float4*)(w1 + o * CH + c0);
        float4 c = *(const float4*)(w2 + o * CH + c0);
        _Float16* d1 = &ldsW1[o * LDA + c0];
        _Float16* d2 = &ldsW2[o * LDA + c0];
        d1[0] = (_Float16)a.x; d1[1] = (_Float16)a.y; d1[2] = (_Float16)a.z; d1[3] = (_Float16)a.w;
        d2[0] = (_Float16)c.x; d2[1] = (_Float16)c.y; d2[2] = (_Float16)c.z; d2[3] = (_Float16)c.w;
    }
    __syncthreads();

    const int wave  = tid >> 5;
    const int lane  = tid & 31;
    const int hl    = lane & 15;
    const int hi    = lane >> 4;          // 0/1 half-wave
    const int mBase = wave * 16;          // this wave's 16-pixel stripe

    v8f acc[8];
    #pragma unroll
    for (int t = 0; t < 8; ++t) acc[t] = (v8f){0,0,0,0,0,0,0,0};

    // ---- GEMM1: h = xn @ w1^T  (M=16 pixels, N=128 out, K=128 in)
    for (int kc = 0; kc < CH; kc += 32) {
        v16h a = load_frag(&ldsA[(mBase + hl) * LDA + kc + hi * 8]);
        #pragma unroll
        for (int t = 0; t < 8; ++t) {
            v16h bf = load_frag(&ldsW1[(t * 16 + hl) * LDA + kc + hi * 8]);
            acc[t] = __builtin_amdgcn_wmma_f32_16x16x32_f16(
                false, a, false, bf, (short)0, acc[t], false, false);
        }
    }

    // ---- bias + ReLU, write H back into this wave's OWN 16 rows of ldsA
    // (only this wave ever reads rows [mBase, mBase+16), so no barrier needed)
    #pragma unroll
    for (int t = 0; t < 8; ++t) {
        const int   n    = t * 16 + hl;
        const float bias = b1[n];
        #pragma unroll
        for (int r = 0; r < 8; ++r) {
            const int m  = mBase + r + hi * 8;     // D layout: M = r + 8*(lane>=16)
            float hval   = acc[t][r] + bias;
            hval         = hval > 0.0f ? hval : 0.0f;
            ldsA[m * LDA + n] = (_Float16)hval;
        }
        acc[t] = (v8f){0,0,0,0,0,0,0,0};
    }

    // ---- GEMM2: ffn = h @ w2^T
    for (int kc = 0; kc < CH; kc += 32) {
        v16h a = load_frag(&ldsA[(mBase + hl) * LDA + kc + hi * 8]);
        #pragma unroll
        for (int t = 0; t < 8; ++t) {
            v16h bf = load_frag(&ldsW2[(t * 16 + hl) * LDA + kc + hi * 8]);
            acc[t] = __builtin_amdgcn_wmma_f32_16x16x32_f16(
                false, a, false, bf, (short)0, acc[t], false, false);
        }
    }

    // ---- transpose accumulators through LDS for coalesced global I/O
    __syncthreads();                      // everyone done reading ldsA/ldsW1
    #pragma unroll
    for (int t = 0; t < 8; ++t) {
        const int n = t * 16 + hl;
        float* orow = &ldsO[n * LDO + mBase + hi * 8];   // 8 contiguous floats
        #pragma unroll
        for (int r = 0; r < 8; ++r) orow[r] = acc[t][r];
    }
    __syncthreads();

    // ---- cooperative epilogue: out = x + ffn + b2, fully coalesced float4
    for (int i = tid; i < CH * (TILE_PIX / 4); i += 256) {   // 4096 float4
        const int c = i >> 5;
        const int p = (i & 31) << 2;
        const size_t row = ((size_t)(b * CH + c)) * HW + pix0;
        float4 xv = *(const float4*)(x + row + p);
        const float bz = b2[c];
        const float* fr = &ldsO[c * LDO + p];
        float4 o;
        o.x = xv.x + fr[0] + bz;
        o.y = xv.y + fr[1] + bz;
        o.z = xv.z + fr[2] + bz;
        o.w = xv.w + fr[3] + bz;
        *(float4*)(out + row + p) = o;
    }
}

// ----------------------------------------------------------------- launch
extern "C" void kernel_launch(void* const* d_in, const int* in_sizes, int n_in,
                              void* d_out, int out_size, void* d_ws, size_t ws_size,
                              hipStream_t stream) {
    const float* x    = (const float*)d_in[0];
    const float* gn_w = (const float*)d_in[1];
    const float* gn_b = (const float*)d_in[2];
    const float* w1   = (const float*)d_in[3];
    const float* b1   = (const float*)d_in[4];
    const float* w2   = (const float*)d_in[5];
    const float* b2   = (const float*)d_in[6];
    float*       out  = (float*)d_out;
    float*       ws   = (float*)d_ws;
    (void)in_sizes; (void)n_in; (void)out_size; (void)ws_size;

    ws_zero_kernel<<<1, 128, 0, stream>>>(ws);
    gn_stats_kernel<<<dim3(64, 64), 256, 0, stream>>>(x, ws);

    const size_t smem = 1024 + (size_t)(TILE_PIX * LDA + 2 * CH * LDA) * sizeof(_Float16);
    fused_gn_ffn_kernel<<<dim3(HW / TILE_PIX, 8), 256, smem, stream>>>(
        x, gn_w, gn_b, w1, b1, w2, b2, ws, out);
}
